// CYP3A4_GNN_48722109006415
// MI455X (gfx1250) — matrix operand order repaired and checked by reading, not compiled, not run on hardware
//
#include <hip/hip_runtime.h>

#define N_NODES  50000
#define N_EDGES  400000
#define E_TOT    450000
#define N_GRAPHS 512
#define FOUT     64
#define NEG_SLOPE 0.2f

typedef __attribute__((ext_vector_type(16))) __bf16 v16bf;
typedef __attribute__((ext_vector_type(8)))  float  v8f;

// ---- monotonic float<->uint encoding for atomicMax on floats ----
__device__ __forceinline__ unsigned encf(float f) {
  unsigned u = __float_as_uint(f);
  return (u & 0x80000000u) ? ~u : (u | 0x80000000u);
}
__device__ __forceinline__ float decf(unsigned v) {
  unsigned u = (v & 0x80000000u) ? (v ^ 0x80000000u) : ~v;
  return __uint_as_float(u);
}

__global__ void zero_u32(unsigned* __restrict__ p, long n) {
  long i = (long)blockIdx.x * blockDim.x + threadIdx.x;
  if (i < n) p[i] = 0u;
}

// Pack W[K,N] fp32 -> Wp[(k/32)*N + n][k%32] bf16 (fragment-native for B loads)
__global__ void pack_w_bf16(const float* __restrict__ W, __bf16* __restrict__ Wp,
                            int K, int N) {
  int i = blockIdx.x * blockDim.x + threadIdx.x;
  if (i >= K * N) return;
  int k = i / N, n = i - k * N;
  Wp[((size_t)(k >> 5) * N + n) * 32 + (k & 31)] = (__bf16)W[i];
}

// ---------------------------------------------------------------------------
// C[M,N] = A[M,K] @ W[K,N] via bf16 WMMA, fp32 accumulate.
// One wave computes a 16 x (16*TN) strip: A fragment loaded once per K-step,
// reused across TN B tiles (packed bf16 layout, contiguous 32B per lane).
// ---------------------------------------------------------------------------
template <int TN>
__global__ void gemm_wmma_bf16(const float* __restrict__ A, const __bf16* __restrict__ Wp,
                               float* __restrict__ C, int M, int K, int N) {
  const int lane    = threadIdx.x & 31;
  const int wave    = threadIdx.x >> 5;
  const int groupsN = N / (16 * TN);
  const int tilesM  = M >> 4;
  int id = blockIdx.x * (blockDim.x >> 5) + wave;
  if (id >= tilesM * groupsN) return;                 // wave-uniform
  const int tm   = id / groupsN;
  const int ng   = id - tm * groupsN;
  const int half = lane >> 4;
  const int l16  = lane & 15;
  const float* arow = A + (size_t)(tm * 16 + l16) * K;
  const int col0 = ng * TN * 16 + l16;                // this lane's B/C column base

  v8f acc[TN];
  #pragma unroll
  for (int t = 0; t < TN; ++t) acc[t] = (v8f){};

  for (int kk = 0; kk < K; kk += 32) {
    // A frag: half 0 -> K kk..kk+15; half 1 -> K kk+16..kk+31 (matches packed B)
    v16bf a;
    const float* ap0 = arow + kk + half * 16;
    #pragma unroll
    for (int i = 0; i < 8; ++i) a[i] = (__bf16)ap0[i];
    #pragma unroll
    for (int i = 0; i < 8; ++i) a[8 + i] = (__bf16)ap0[8 + i];
    // B frags: lanes 0-15 hold K=kk..kk+15, lanes 16-31 hold K=kk+16..kk+31
    const __bf16* wbase = Wp + ((size_t)(kk >> 5) * N + col0) * 32 + half * 16;
    #pragma unroll
    for (int t = 0; t < TN; ++t) {
      v16bf b = *(const v16bf*)(wbase + (size_t)t * (16 * 32));
      acc[t] = __builtin_amdgcn_wmma_f32_16x16x32_bf16(false, a, false, b,
                                                       (short)0, acc[t], false, false);
    }
  }
  // C/D layout: VGPR r -> M = r + 8*half, N = lane%16
  #pragma unroll
  for (int t = 0; t < TN; ++t) {
    float* crow = C + (size_t)(tm * 16 + half * 8) * N + col0 + t * 16;
    #pragma unroll
    for (int r = 0; r < 8; ++r) crow[(size_t)r * N] = acc[t][r];
  }
}

// ---------------------------------------------------------------------------
// a_s[n,h] = sum_c h[n,h,c]*a_src[h,c] ; same for a_d
// ---------------------------------------------------------------------------
__global__ void node_attn(const float* __restrict__ h, const float* __restrict__ a_src,
                          const float* __restrict__ a_dst, float* __restrict__ as,
                          float* __restrict__ ad, int H) {
  int i = blockIdx.x * blockDim.x + threadIdx.x;
  if (i >= N_NODES * H) return;
  int hd = i % H;
  const float* hp = h + (size_t)(i / H) * (H * FOUT) + hd * FOUT;
  const float* sp = a_src + hd * FOUT;
  const float* dp = a_dst + hd * FOUT;
  float sa = 0.f, da = 0.f;
  #pragma unroll 4
  for (int c = 0; c < FOUT; ++c) { float v = hp[c]; sa += v * sp[c]; da += v * dp[c]; }
  as[i] = sa; ad[i] = da;
}

__device__ __forceinline__ void edge_sd(const int* __restrict__ ei, int e, int& s, int& d) {
  if (e < N_EDGES) { s = ei[e]; d = ei[N_EDGES + e]; }
  else             { s = d = e - N_EDGES; }           // self-loops
}

// pass 1: e = leaky_relu(a_s[src]+a_d[dst]); segment max over dst
__global__ void edge_logits(const int* __restrict__ ei, const float* __restrict__ as,
                            const float* __restrict__ ad, float* __restrict__ ebuf,
                            unsigned* __restrict__ menc, int H) {
  int i = blockIdx.x * blockDim.x + threadIdx.x;
  if (i >= E_TOT * H) return;
  int e = i / H, hd = i % H, s, d;
  edge_sd(ei, e, s, d);
  float v = as[s * H + hd] + ad[d * H + hd];
  v = v > 0.f ? v : NEG_SLOPE * v;
  ebuf[i] = v;
  atomicMax(&menc[d * H + hd], encf(v));
}

// pass 2: e = exp(e - m[dst]); segment sum over dst
__global__ void edge_exp(const int* __restrict__ ei, float* __restrict__ ebuf,
                         const unsigned* __restrict__ menc, float* __restrict__ denom, int H) {
  int i = blockIdx.x * blockDim.x + threadIdx.x;
  if (i >= E_TOT * H) return;
  int e = i / H, hd = i % H, s, d;
  edge_sd(ei, e, s, d);
  float a = expf(ebuf[i] - decf(menc[d * H + hd]));
  ebuf[i] = a;
  atomicAdd(&denom[d * H + hd], a);
}

// pass 3: out[dst,h,:] += h[src,h,:] * alpha
__global__ void edge_aggr(const int* __restrict__ ei, const float* __restrict__ ebuf,
                          const float* __restrict__ denom, const float* __restrict__ h,
                          float* __restrict__ out, int H) {
  int i = blockIdx.x * blockDim.x + threadIdx.x;
  if (i >= E_TOT * H) return;
  int e = i / H, hd = i % H, s, d;
  edge_sd(ei, e, s, d);
  float alpha = ebuf[i] / (denom[d * H + hd] + 1e-16f);
  const float* hp = h + (size_t)s * (H * FOUT) + hd * FOUT;
  float* op = out + (size_t)d * (H * FOUT) + hd * FOUT;
  #pragma unroll 8
  for (int c = 0; c < FOUT; ++c) atomicAdd(&op[c], hp[c] * alpha);
}

// X[r,c] += bias[c] (in-place), accumulate per-channel sum & sumsq
__global__ void bias_bnstats(float* __restrict__ X, const float* __restrict__ bias,
                             float* __restrict__ sum, float* __restrict__ sumsq,
                             int rows, int width, int rowsPerBlock) {
  int c = threadIdx.x;
  if (c >= width) return;
  int r0 = blockIdx.x * rowsPerBlock;
  int r1 = min(r0 + rowsPerBlock, rows);
  float b = bias ? bias[c] : 0.f;
  float s = 0.f, ss = 0.f;
  for (int r = r0; r < r1; ++r) {
    float v = X[(size_t)r * width + c] + b;
    X[(size_t)r * width + c] = v;
    s += v; ss += v * v;
  }
  atomicAdd(&sum[c], s);
  atomicAdd(&sumsq[c], ss);
}

__global__ void bn_apply(const float* __restrict__ X, float* __restrict__ Y,
                         const float* __restrict__ sum, const float* __restrict__ sumsq,
                         const float* __restrict__ gamma, const float* __restrict__ beta,
                         long n, int width, float invn, int do_relu) {
  long i = (long)blockIdx.x * blockDim.x + threadIdx.x;
  if (i >= n) return;
  int c = (int)(i % width);
  float mu  = sum[c] * invn;
  float var = sumsq[c] * invn - mu * mu;
  float v = (X[i] - mu) * rsqrtf(var + 1e-5f) * gamma[c] + beta[c];
  if (do_relu) v = fmaxf(v, 0.f);
  Y[i] = v;
}

// pooling scatter: per (node, channel)
__global__ void pool_scatter(const float* __restrict__ h, const int* __restrict__ batch,
                             float* __restrict__ cnt, float* __restrict__ xsum,
                             unsigned* __restrict__ xmax) {
  int i = blockIdx.x * blockDim.x + threadIdx.x;
  if (i >= N_NODES * FOUT) return;
  int node = i / FOUT, c = i - node * FOUT;
  int b = batch[node];
  float v = h[i];
  atomicAdd(&xsum[b * FOUT + c], v);
  atomicMax(&xmax[b * FOUT + c], encf(v));
  if (c == 0) atomicAdd(&cnt[b], 1.f);
}

// z[g,0:64]=mean, z[g,64:128]=max   (z row stride = 160)
__global__ void pool_finalize(const float* __restrict__ cnt, const float* __restrict__ xsum,
                              const unsigned* __restrict__ xmax, float* __restrict__ z) {
  int i = blockIdx.x * blockDim.x + threadIdx.x;
  if (i >= N_GRAPHS * FOUT) return;
  int g = i / FOUT, c = i - g * FOUT;
  float n = cnt[g];
  float mean = xsum[i] / fmaxf(n, 1.f);
  float mx = (n > 0.f) ? decf(xmax[i]) : 0.f;
  z[(size_t)g * 160 + c] = mean;
  z[(size_t)g * 160 + 64 + c] = mx;
}

// global-feature MLP: relu(g@W1+b1)@W2+b2 -> z[g,128:160]
__global__ void gmlp_kernel(const float* __restrict__ G, const float* __restrict__ W1,
                            const float* __restrict__ b1, const float* __restrict__ W2,
                            const float* __restrict__ b2, float* __restrict__ z) {
  __shared__ float h1[32];
  int g = blockIdx.x, j = threadIdx.x;
  float acc = b1[j];
  for (int k = 0; k < 6; ++k) acc += G[g * 6 + k] * W1[k * 32 + j];
  h1[j] = fmaxf(acc, 0.f);
  __syncthreads();
  float acc2 = b2[j];
  for (int k = 0; k < 32; ++k) acc2 += h1[k] * W2[k * 32 + j];
  z[(size_t)g * 160 + 128 + j] = acc2;
}

// generic small FC: out[g,j] = (relu)(Z[g,:]@W[:,j] + b[j]); one block per row
__global__ void mlp_fc(const float* __restrict__ Z, const float* __restrict__ W,
                       const float* __restrict__ b, float* __restrict__ out,
                       int K, int N, int do_relu) {
  extern __shared__ float zs[];
  int g = blockIdx.x, j = threadIdx.x;
  for (int k = j; k < K; k += blockDim.x) zs[k] = Z[(size_t)g * K + k];
  __syncthreads();
  float acc = b[j];
  for (int k = 0; k < K; ++k) acc += zs[k] * W[(size_t)k * N + j];
  if (do_relu) acc = fmaxf(acc, 0.f);
  out[(size_t)g * N + j] = acc;
}

__global__ void pred_final(const float* __restrict__ p2, const float* __restrict__ W3,
                           const float* __restrict__ b3, float* __restrict__ out, int K) {
  int g = blockIdx.x * blockDim.x + threadIdx.x;
  if (g >= N_GRAPHS) return;
  float acc = b3[0];
  for (int k = 0; k < K; ++k) acc += p2[(size_t)g * K + k] * W3[k];
  out[g] = acc;
}

// ---------------------------------------------------------------------------
extern "C" void kernel_launch(void* const* d_in, const int* in_sizes, int n_in,
                              void* d_out, int out_size, void* d_ws, size_t ws_size,
                              hipStream_t stream) {
  (void)in_sizes; (void)n_in; (void)out_size; (void)ws_size;

  const float* x     = (const float*)d_in[0];
  const int*   ei    = (const int*)d_in[1];
  const int*   batch = (const int*)d_in[2];
  const float* gfeat = (const float*)d_in[3];

  struct LayerP { const float *W, *a_src, *a_dst, *b, *gamma, *beta; };
  LayerP L[4];
  int p = 4;
  for (int l = 0; l < 4; ++l) {
    L[l].W = (const float*)d_in[p]; L[l].a_src = (const float*)d_in[p+1];
    L[l].a_dst = (const float*)d_in[p+2]; L[l].b = (const float*)d_in[p+3];
    L[l].gamma = (const float*)d_in[p+4]; L[l].beta = (const float*)d_in[p+5];
    p += 6;
  }
  const float* gW1 = (const float*)d_in[p+0]; const float* gb1 = (const float*)d_in[p+1];
  const float* gW2 = (const float*)d_in[p+2]; const float* gb2 = (const float*)d_in[p+3];
  const float* pW1 = (const float*)d_in[p+4]; const float* pb1 = (const float*)d_in[p+5];
  const float* pga = (const float*)d_in[p+6]; const float* pbe = (const float*)d_in[p+7];
  const float* pW2 = (const float*)d_in[p+8]; const float* pb2 = (const float*)d_in[p+9];
  const float* pW3 = (const float*)d_in[p+10]; const float* pb3 = (const float*)d_in[p+11];

  const int fin[4]   = {64, 256, 256, 256};
  const int heads[4] = {4, 4, 4, 1};

  // ---- workspace carve (256B aligned) ----
  char* w = (char*)d_ws;
  size_t off = 0;
  auto carve = [&](size_t bytes) { void* r = w + off; off = (off + bytes + 255) & ~(size_t)255; return r; };
  float*    bufA  = (float*)carve((size_t)N_NODES * 256 * 4);   // activations
  float*    bufB  = (float*)carve((size_t)N_NODES * 256 * 4);   // H = X@W
  float*    bufC  = (float*)carve((size_t)N_NODES * 256 * 4);   // aggregated
  __bf16*   Wp    = (__bf16*)carve((size_t)256 * 256 * 2);      // packed bf16 weights
  float*    a_s   = (float*)carve((size_t)N_NODES * 4 * 4);
  float*    a_d   = (float*)carve((size_t)N_NODES * 4 * 4);
  unsigned* menc  = (unsigned*)carve((size_t)N_NODES * 4 * 4);
  float*    denom = (float*)carve((size_t)N_NODES * 4 * 4);
  float*    ebuf  = (float*)carve((size_t)E_TOT * 4 * 4);
  float*    bnsum = (float*)carve(2 * 256 * 4);                 // sum | sumsq adjacent
  float*    bnsq  = bnsum + 256;
  float*    cnt   = (float*)carve(N_GRAPHS * 4);
  float*    xsum  = (float*)carve((size_t)N_GRAPHS * FOUT * 4);
  unsigned* xmax  = (unsigned*)carve((size_t)N_GRAPHS * FOUT * 4);
  float*    zbuf  = (float*)carve((size_t)N_GRAPHS * 160 * 4);
  float*    p1    = (float*)carve((size_t)N_GRAPHS * 256 * 4);
  float*    p2    = (float*)carve((size_t)N_GRAPHS * 128 * 4);

  auto zero = [&](void* ptr, long n) {
    zero_u32<<<(unsigned)((n + 255) / 256), 256, 0, stream>>>((unsigned*)ptr, n);
  };

  const float* in_ptr = x;
  for (int l = 0; l < 4; ++l) {
    const int H = heads[l];
    const int width = H * FOUT;
    // 1) pack W to bf16 fragment layout, then H = X @ W via WMMA strips
    pack_w_bf16<<<(fin[l] * width + 255) / 256, 256, 0, stream>>>(L[l].W, Wp, fin[l], width);
    if (width == 256) {      // TN=8: 2 n-groups, A read twice
      int waves = (N_NODES / 16) * 2;
      gemm_wmma_bf16<8><<<(waves + 7) / 8, 256, 0, stream>>>(in_ptr, Wp, bufB,
                                                             N_NODES, fin[l], width);
    } else {                 // TN=4: 1 n-group, A read once
      int waves = (N_NODES / 16);
      gemm_wmma_bf16<4><<<(waves + 7) / 8, 256, 0, stream>>>(in_ptr, Wp, bufB,
                                                             N_NODES, fin[l], width);
    }
    // 2) zero accumulators
    zero(bufC, (long)N_NODES * width);
    zero(menc, (long)N_NODES * H);
    zero(denom, (long)N_NODES * H);
    // 3) attention coefficients per node
    node_attn<<<(N_NODES * H + 255) / 256, 256, 0, stream>>>(bufB, L[l].a_src, L[l].a_dst,
                                                             a_s, a_d, H);
    // 4-6) edge passes: logits+segmax, exp+segsum, weighted scatter-add
    int eg = (E_TOT * H + 255) / 256;
    edge_logits<<<eg, 256, 0, stream>>>(ei, a_s, a_d, ebuf, menc, H);
    edge_exp<<<eg, 256, 0, stream>>>(ei, ebuf, menc, denom, H);
    edge_aggr<<<eg, 256, 0, stream>>>(ei, ebuf, denom, bufB, bufC, H);
    // 7-9) bias + BN stats, BN apply + ReLU -> bufA
    zero(bnsum, 2 * width);
    bias_bnstats<<<(N_NODES + 255) / 256, width, 0, stream>>>(bufC, L[l].b, bnsum, bnsq,
                                                              N_NODES, width, 256);
    long n = (long)N_NODES * width;
    bn_apply<<<(unsigned)((n + 255) / 256), 256, 0, stream>>>(bufC, bufA, bnsum, bnsq,
        L[l].gamma, L[l].beta, n, width, 1.f / N_NODES, 1);
    in_ptr = bufA;
  }

  // ---- pooling over graphs ----
  zero(cnt, N_GRAPHS);
  zero(xsum, (long)N_GRAPHS * FOUT);
  zero(xmax, (long)N_GRAPHS * FOUT);
  pool_scatter<<<(N_NODES * FOUT + 255) / 256, 256, 0, stream>>>(bufA, batch, cnt, xsum, xmax);
  pool_finalize<<<(N_GRAPHS * FOUT + 255) / 256, 256, 0, stream>>>(cnt, xsum, xmax, zbuf);
  gmlp_kernel<<<N_GRAPHS, 32, 0, stream>>>(gfeat, gW1, gb1, gW2, gb2, zbuf);

  // ---- prediction head ----
  mlp_fc<<<N_GRAPHS, 256, 160 * 4, stream>>>(zbuf, pW1, pb1, p1, 160, 256, 1);
  zero(bnsum, 2 * 256);
  bias_bnstats<<<(N_GRAPHS + 63) / 64, 256, 0, stream>>>(p1, nullptr, bnsum, bnsq,
                                                         N_GRAPHS, 256, 64);
  long np = (long)N_GRAPHS * 256;
  bn_apply<<<(unsigned)((np + 255) / 256), 256, 0, stream>>>(p1, p1, bnsum, bnsq,
      pga, pbe, np, 256, 1.f / N_GRAPHS, 0);
  mlp_fc<<<N_GRAPHS, 128, 256 * 4, stream>>>(p1, pW2, pb2, p2, 256, 128, 1);
  pred_final<<<(N_GRAPHS + 255) / 256, 256, 0, stream>>>(p2, pW3, pb3, (float*)d_out, 128);
}
